// TrueQuantizedLinear_54580444397815
// MI455X (gfx1250) — matrix operand order repaired and checked by reading, not compiled, not run on hardware
//
#include <hip/hip_runtime.h>

typedef __attribute__((ext_vector_type(16))) _Float16 v16h;
typedef __attribute__((ext_vector_type(8)))  _Float16 v8h;
typedef __attribute__((ext_vector_type(8)))  float    v8f;

#define IN_F   4096
#define OUT_F  4096
#define M_TOT  16384   // 8 * 2048

#define BLK_M  128
#define BLK_N  256
#define BLK_K  64
#define LDSTR  72      // padded LDS row stride in halves (144 B = 36 banks, 16B aligned)

__constant__ float NF4_LUT[16] = {
    -1.0f, -0.6961928009986877f, -0.5250730514526367f, -0.39491748809814453f,
    -0.28444138169288635f, -0.18477343022823334f, -0.09105003625154495f, 0.0f,
    0.07958029955625534f, 0.16093020141124725f, 0.24611230194568634f,
    0.33791524171829224f, 0.44070982933044434f, 0.5626170039176941f,
    0.7229568362236023f, 1.0f};

// ---------------------------------------------------------------------------
// Kernel 1: NF4 -> fp16 dequant. One thread per packed byte -> two halves,
// written as one 32-bit store (coalesced). W_fp16 is [OUT_F][IN_F] row-major.
// ---------------------------------------------------------------------------
__global__ __launch_bounds__(256)
void dequant_nf4(const int* __restrict__ wp, const float* __restrict__ scale,
                 _Float16* __restrict__ Wh, int n_packed) {
    int t = blockIdx.x * blockDim.x + threadIdx.x;
    if (t >= n_packed) return;
    float s = scale[0];
    unsigned b = (unsigned)wp[t] & 0xFFu;
    union { _Float16 h[2]; unsigned u; } pk;
    pk.h[0] = (_Float16)(NF4_LUT[b & 15u] * s);          // element 2i   (low nibble)
    pk.h[1] = (_Float16)(NF4_LUT[(b >> 4) & 15u] * s);   // element 2i+1 (high nibble)
    ((unsigned*)Wh)[t] = pk.u;
}

// ---------------------------------------------------------------------------
// Kernel 2: x fp32 -> fp16. 4 elements per thread (float4 in, 8B out).
// ---------------------------------------------------------------------------
__global__ __launch_bounds__(256)
void cvt_x_f16(const float* __restrict__ x, _Float16* __restrict__ Xh, long n4) {
    long t = (long)blockIdx.x * blockDim.x + threadIdx.x;
    if (t >= n4) return;
    float4 v = ((const float4*)x)[t];
    union { _Float16 h[4]; unsigned long long u; } pk;
    pk.h[0] = (_Float16)v.x; pk.h[1] = (_Float16)v.y;
    pk.h[2] = (_Float16)v.z; pk.h[3] = (_Float16)v.w;
    ((unsigned long long*)Xh)[t] = pk.u;
}

// ---------------------------------------------------------------------------
// Helpers: CDNA5 async global->LDS copy (no intermediate VGPRs, ASYNCcnt).
// LDS address = low 32 bits of the generic pointer (ISA aperture rule).
// ---------------------------------------------------------------------------
__device__ __forceinline__ void async_copy_b128(const _Float16* gsrc, _Float16* ldst) {
    unsigned loff = (unsigned)(unsigned long long)ldst;
    unsigned long long ga = (unsigned long long)gsrc;
    asm volatile("global_load_async_to_lds_b128 %0, %1, off"
                 :: "v"(loff), "v"(ga) : "memory");
}
__device__ __forceinline__ void wait_async0() {
    asm volatile("s_wait_asynccnt 0x0" ::: "memory");
}

// ---------------------------------------------------------------------------
// Kernel 3: GEMM  out[M][N] = Xh[M][K] * Wh[N][K]^T + bias   via WMMA f16.
// Block = 256 threads = 8 waves (wave32), wave grid 2(M) x 4(N).
// Block tile 128x256, wave tile 64x64 => 4x4 accumulators 16x16.
// K staged 64-deep through double-buffered LDS filled by async-to-LDS loads.
// Per stage per wave: 32 x v_wmma_f32_16x16x32_f16, 32 x ds_load_b128.
// ---------------------------------------------------------------------------
__global__ __launch_bounds__(256)
void gemm_wmma_f16(const _Float16* __restrict__ Xh,   // [M_TOT][IN_F]
                   const _Float16* __restrict__ Wh,   // [OUT_F][IN_F]
                   const float*    __restrict__ bias, // [OUT_F]
                   float*          __restrict__ out)  // [M_TOT][OUT_F]
{
    __shared__ _Float16 lA[2][BLK_M * LDSTR];   // 2 x 18 KB
    __shared__ _Float16 lB[2][BLK_N * LDSTR];   // 2 x 36 KB

    const int tid  = threadIdx.x;
    const int lane = tid & 31;
    const int wave = tid >> 5;         // 0..7
    const int wm0  = (wave & 1) * 64;  // wave M offset within block tile
    const int wn0  = (wave >> 1) * 64; // wave N offset within block tile
    const int lhalf = lane >> 4;       // 0: lanes 0-15, 1: lanes 16-31

    const int m0 = blockIdx.y * BLK_M;
    const int n0 = blockIdx.x * BLK_N;

    v8f acc[4][4];
#pragma unroll
    for (int i = 0; i < 4; ++i)
#pragma unroll
        for (int j = 0; j < 4; ++j)
            acc[i][j] = (v8f){0.f, 0.f, 0.f, 0.f, 0.f, 0.f, 0.f, 0.f};

    // ---- async stage of one 64-deep K slice into LDS buffer s ----
    // A tile: 128 rows x 64 halves: thread t -> row t>>1, 32-half segment t&1.
    // B tile: 256 rows x 64 halves: thread t -> row t, whole 64-half row.
    auto issue_stage = [&](int k0, int s) {
        {
            const int r  = tid >> 1;
            const int hb = (tid & 1) * 32;
            const _Float16* g = &Xh[(long)(m0 + r) * IN_F + k0 + hb];
            _Float16* l = &lA[s][r * LDSTR + hb];
#pragma unroll
            for (int c = 0; c < 4; ++c)
                async_copy_b128(g + c * 8, l + c * 8);
        }
        {
            const _Float16* g = &Wh[(long)(n0 + tid) * IN_F + k0];
            _Float16* l = &lB[s][tid * LDSTR];
#pragma unroll
            for (int c = 0; c < 8; ++c)
                async_copy_b128(g + c * 8, l + c * 8);
        }
    };

    issue_stage(0, 0);

    int s = 0;
#pragma unroll 1
    for (int k0 = 0; k0 < IN_F; k0 += BLK_K, s ^= 1) {
        wait_async0();      // my stage-k0 async loads have landed in LDS
        __syncthreads();    // everyone's have; prior stage fully consumed

        if (k0 + BLK_K < IN_F)
            issue_stage(k0 + BLK_K, s ^ 1);   // overlap with compute below

#pragma unroll
        for (int kq = 0; kq < 2; ++kq) {
            // B fragments: lane<16 holds K 0..15 of col N=lane; lane>=16 K 16..31
            union { v16h v; v8h h[2]; } Bf[4];
#pragma unroll
            for (int nt = 0; nt < 4; ++nt) {
                const _Float16* pb =
                    &lB[s][(wn0 + nt * 16 + (lane & 15)) * LDSTR + kq * 32 + lhalf * 16];
                Bf[nt].h[0] = *(const v8h*)(pb);
                Bf[nt].h[1] = *(const v8h*)(pb + 8);
            }
            // A fragments: lane<16 holds K 0..15; lane>=16 K 16..31
#pragma unroll
            for (int mt = 0; mt < 4; ++mt) {
                union { v16h v; v8h h[2]; } Af;
                const _Float16* pa =
                    &lA[s][(wm0 + mt * 16 + (lane & 15)) * LDSTR + kq * 32 + lhalf * 16];
                Af.h[0] = *(const v8h*)(pa);
                Af.h[1] = *(const v8h*)(pa + 8);
#pragma unroll
                for (int nt = 0; nt < 4; ++nt)
                    acc[mt][nt] = __builtin_amdgcn_wmma_f32_16x16x32_f16(
                        false, Af.v, false, Bf[nt].v,
                        (short)0, acc[mt][nt], false, false);
            }
        }
        // no trailing barrier: top-of-loop wait+barrier orders buffer reuse
    }

    // ---- epilogue: bias add + store. C/D layout: VGPR r -> M = r + 8*(lane>=16),
    //      N = lane & 15. ----
#pragma unroll
    for (int nt = 0; nt < 4; ++nt) {
        const int col = n0 + wn0 + nt * 16 + (lane & 15);
        const float bz = bias[col];
#pragma unroll
        for (int mt = 0; mt < 4; ++mt) {
            const int rowBase = m0 + wm0 + mt * 16 + lhalf * 8;
#pragma unroll
            for (int r = 0; r < 8; ++r)
                out[(long)(rowBase + r) * OUT_F + col] = acc[mt][nt][r] + bz;
        }
    }
}

// ---------------------------------------------------------------------------
extern "C" void kernel_launch(void* const* d_in, const int* in_sizes, int n_in,
                              void* d_out, int out_size, void* d_ws, size_t ws_size,
                              hipStream_t stream) {
    const float* x        = (const float*)d_in[0];   // [8,2048,4096] fp32
    const int*   w_packed = (const int*)  d_in[1];   // [OUT_F*IN_F/2]
    const float* scale    = (const float*)d_in[2];   // [1]
    const float* bias     = (const float*)d_in[3];   // [OUT_F]
    float*       outp     = (float*)d_out;           // [8,2048,4096] fp32

    // workspace partition: Xh (fp16, 128 MB) then Wh (fp16, 32 MB)
    _Float16* Xh = (_Float16*)d_ws;
    _Float16* Wh = (_Float16*)((char*)d_ws + (size_t)M_TOT * IN_F * sizeof(_Float16));

    // 1) dequant W: one thread per packed byte
    {
        int n_packed = OUT_F * IN_F / 2;            // 8,388,608
        dim3 grid((n_packed + 255) / 256), block(256);
        dequant_nf4<<<grid, block, 0, stream>>>(w_packed, scale, Wh, n_packed);
    }
    // 2) convert x to fp16: 4 elements/thread
    {
        long n4 = (long)M_TOT * IN_F / 4;           // 16,777,216
        dim3 grid((unsigned)((n4 + 255) / 256)), block(256);
        cvt_x_f16<<<grid, block, 0, stream>>>(x, Xh, n4);
    }
    // 3) GEMM + bias
    {
        dim3 grid(OUT_F / BLK_N, M_TOT / BLK_M);    // (16, 128)
        dim3 block(256);
        gemm_wmma_f16<<<grid, block, 0, stream>>>(Xh, Wh, bias, outp);
    }
}
